// LuongAttension_266287972944
// MI455X (gfx1250) — matrix operand order repaired and checked by reading, not compiled, not run on hardware
//
#include <hip/hip_runtime.h>
#include <hip/hip_bf16.h>

// ---------------------------------------------------------------------------
// Luong attention on gfx1250 (MI455X): 3 batched GEMMs via v_wmma_f32_16x16x32_f16
// (f32 inputs staged to LDS as f16, f32 accumulation) + row softmax.
// ---------------------------------------------------------------------------

typedef _Float16 v16h __attribute__((ext_vector_type(16)));
typedef _Float16 v8h  __attribute__((ext_vector_type(8)));
typedef _Float16 v4h  __attribute__((ext_vector_type(4)));
typedef float    v8f  __attribute__((ext_vector_type(8)));

union F16x16 {
    v16h v;
    struct { v8h lo, hi; } p;
};

// Tile configuration
constexpr int BM = 128;          // block tile M
constexpr int BN = 128;          // block tile N
constexpr int BK = 32;           // K step (matches WMMA K for f16)
constexpr int LPAD = 8;          // halves of padding per LDS row
constexpr int LDSL = BK + LPAD;  // 40 halves = 80 bytes/row (16B aligned)

// ---------------------------------------------------------------------------
// Generic tiled WMMA GEMM:  Out[M,N] = A[M,K] (f32) * B (+ bias)
//   B_TRANS == false : B source is row-major [K, N], dtype TB
//   B_TRANS == true  : B source is row-major [N, K], dtype TB (f16 keys)
// Output f16 (OUT_HALF) or f32.
// Block: 256 threads = 8 waves (4 along M x 2 along N); wave tile 32x64;
// per-wave 2x4 accumulators of 16x16.
// ---------------------------------------------------------------------------
template <typename TB, bool B_TRANS, bool ADD_BIAS, bool OUT_HALF>
__global__ __launch_bounds__(256)
void gemm_wmma_f16(const float* __restrict__ Aall,
                   const TB*    __restrict__ Ball,
                   const float* __restrict__ bias,
                   void*        __restrict__ Out,
                   int K, int lda, int ldb, int ldc,
                   long long strideA, long long strideB, long long strideC)
{
    __shared__ _Float16 ldsA[BM * LDSL];   // [m][k] tiles, f16
    __shared__ _Float16 ldsB[BN * LDSL];   // [n][k] tiles, f16

    const int tid  = threadIdx.x;
    const int lane = tid & 31;
    const int wave = tid >> 5;
    const int wm   = wave & 3;     // wave row   (0..3) -> 32 rows each
    const int wn   = wave >> 2;    // wave col   (0..1) -> 64 cols each
    const int m16  = lane & 15;
    const int hi   = lane >> 4;    // half-wave selector

    const int bm0 = blockIdx.y * BM;
    const int bn0 = blockIdx.x * BN;

    const float* Ag = Aall + (long long)blockIdx.z * strideA;
    const TB*    Bg = Ball + (long long)blockIdx.z * strideB;
    const long long obase = (long long)blockIdx.z * strideC;

    v8f acc[2][4] = {};

    for (int k0 = 0; k0 < K; k0 += BK) {
        // ---- stage A tile: BM x BK f32 -> f16 (float4 coalesced reads) ----
        #pragma unroll
        for (int it = 0; it < 4; ++it) {
            const int idx = tid + it * 256;      // 0..1023
            const int r   = idx >> 3;            // row in tile
            const int c   = (idx & 7) << 2;      // col (multiple of 4)
            const float4 f = *(const float4*)(Ag + (long long)(bm0 + r) * lda + k0 + c);
            v4h h = { (_Float16)f.x, (_Float16)f.y, (_Float16)f.z, (_Float16)f.w };
            *(v4h*)&ldsA[r * LDSL + c] = h;
        }
        // ---- stage B tile into ldsB[n][k] ----
        if constexpr (B_TRANS) {
            // keys case: B source is f16 [N, K] row-major; contiguous along k.
            #pragma unroll
            for (int it = 0; it < 2; ++it) {
                const int idx = tid + it * 256;  // 0..511
                const int n   = idx >> 2;
                const int ck  = (idx & 3) << 3;  // 8-half chunks
                const v8h v = *(const v8h*)(Bg + (long long)(bn0 + n) * ldb + k0 + ck);
                *(v8h*)&ldsB[n * LDSL + ck] = v;
            }
        } else {
            // W / values case: B source is f32 [K, N] row-major; read contiguous
            // along n (coalesced), scatter-transpose into LDS.
            #pragma unroll
            for (int it = 0; it < 4; ++it) {
                const int idx = tid + it * 256;  // 0..1023
                const int kk  = idx >> 5;
                const int c   = (idx & 31) << 2;
                const float4 f = *(const float4*)(Bg + (long long)(k0 + kk) * ldb + bn0 + c);
                ldsB[(c + 0) * LDSL + kk] = (_Float16)f.x;
                ldsB[(c + 1) * LDSL + kk] = (_Float16)f.y;
                ldsB[(c + 2) * LDSL + kk] = (_Float16)f.z;
                ldsB[(c + 3) * LDSL + kk] = (_Float16)f.w;
            }
        }
        __syncthreads();

        // ---- load fragments per ISA 7.12.2 layouts ----
        // A 16x32: lane m = lane&15; halves K = hi*8+{0..7} and 16+hi*8+{0..7}
        F16x16 a[2];
        #pragma unroll
        for (int i = 0; i < 2; ++i) {
            const _Float16* p = &ldsA[(wm * 32 + i * 16 + m16) * LDSL];
            a[i].p.lo = *(const v8h*)(p + hi * 8);
            a[i].p.hi = *(const v8h*)(p + 16 + hi * 8);
        }
        // B 32x16: lane n = lane&15; halves K = hi*8+{0..7} and 16+hi*8+{0..7}
        F16x16 b[4];
        #pragma unroll
        for (int j = 0; j < 4; ++j) {
            const _Float16* p = &ldsB[(wn * 64 + j * 16 + m16) * LDSL];
            b[j].p.lo = *(const v8h*)(p + hi * 8);
            b[j].p.hi = *(const v8h*)(p + 16 + hi * 8);
        }

        #pragma unroll
        for (int i = 0; i < 2; ++i)
            #pragma unroll
            for (int j = 0; j < 4; ++j)
                acc[i][j] = __builtin_amdgcn_wmma_f32_16x16x32_f16(
                    /*neg_a=*/false, a[i].v, /*neg_b=*/false, b[j].v,
                    /*c_mod=*/(short)0, acc[i][j],
                    /*reuse_a=*/false, /*reuse_b=*/false);
        __syncthreads();
    }

    // ---- epilogue: C/D layout: lanes 0-15 N=lane, VGPR r -> M=r; lanes 16-31 -> M=8+r
    #pragma unroll
    for (int i = 0; i < 2; ++i) {
        const int row = bm0 + wm * 32 + i * 16 + hi * 8;
        #pragma unroll
        for (int j = 0; j < 4; ++j) {
            const int col = bn0 + wn * 64 + j * 16 + m16;
            float bz = 0.0f;
            if constexpr (ADD_BIAS) bz = bias[col];
            #pragma unroll
            for (int r = 0; r < 8; ++r) {
                const float v = acc[i][j][r] + bz;
                const long long o = obase + (long long)(row + r) * ldc + col;
                if constexpr (OUT_HALF) ((_Float16*)Out)[o] = (_Float16)v;
                else                    ((float*)Out)[o]    = v;
            }
        }
    }
}

// ---------------------------------------------------------------------------
// In-place row softmax, 1 block (256 threads, 8 waves) per row of 2048.
// wave32: shuffle-reduce within 32 lanes, then LDS across 8 waves.
// ---------------------------------------------------------------------------
__global__ __launch_bounds__(256)
void softmax_rows_2048(float* __restrict__ data, int ncols)
{
    const long long row = blockIdx.x;
    float* p = data + row * (long long)ncols;
    const int tid = threadIdx.x;

    float v[8];
    float m = -3.4e38f;
    #pragma unroll
    for (int i = 0; i < 8; ++i) {
        v[i] = p[tid + i * 256];
        m = fmaxf(m, v[i]);
    }
    #pragma unroll
    for (int off = 16; off > 0; off >>= 1)
        m = fmaxf(m, __shfl_xor(m, off, 32));

    __shared__ float redm[8];
    __shared__ float reds[8];
    if ((tid & 31) == 0) redm[tid >> 5] = m;
    __syncthreads();
    m = redm[0];
    #pragma unroll
    for (int w = 1; w < 8; ++w) m = fmaxf(m, redm[w]);

    float s = 0.0f;
    #pragma unroll
    for (int i = 0; i < 8; ++i) {
        v[i] = __expf(v[i] - m);
        s += v[i];
    }
    #pragma unroll
    for (int off = 16; off > 0; off >>= 1)
        s += __shfl_xor(s, off, 32);
    if ((tid & 31) == 0) reds[tid >> 5] = s;
    __syncthreads();
    s = reds[0];
    #pragma unroll
    for (int w = 1; w < 8; ++w) s += reds[w];

    const float inv = 1.0f / s;
    #pragma unroll
    for (int i = 0; i < 8; ++i) p[tid + i * 256] = v[i] * inv;
}

// ---------------------------------------------------------------------------
extern "C" void kernel_launch(void* const* d_in, const int* in_sizes, int n_in,
                              void* d_out, int out_size, void* d_ws, size_t ws_size,
                              hipStream_t stream)
{
    (void)in_sizes; (void)n_in; (void)out_size; (void)ws_size;

    constexpr int  Bb = 16, TQ = 2048, TV = 2048, Dd = 1024, UN = 1024;

    const float* query  = (const float*)d_in[0];   // [B, TQ, UN]
    const float* values = (const float*)d_in[1];   // [B, TV, Dd]
    const float* Wk     = (const float*)d_in[2];   // [Dd, UN]
    const float* Wb     = (const float*)d_in[3];   // [UN]

    float* ctx   = (float*)d_out;                                  // [B, TQ, Dd]
    float* align = ctx + (long long)Bb * TQ * Dd;                  // [B, TQ, TV]
    _Float16* keys = (_Float16*)d_ws;                              // [B, TV, UN] f16 (64 MiB)

    const dim3 blk(256);

    // 1) keys = values @ W + bias  (f32 -> f16 workspace)
    gemm_wmma_f16<float, /*B_TRANS=*/false, /*BIAS=*/true, /*OUT_HALF=*/true>
        <<<dim3(UN / BN, TV / BM, Bb), blk, 0, stream>>>(
            values, Wk, Wb, keys,
            /*K=*/Dd, /*lda=*/Dd, /*ldb=*/UN, /*ldc=*/UN,
            (long long)TV * Dd, 0LL, (long long)TV * UN);

    // 2) score = query @ keys^T  (f32 into alignment slot of d_out)
    gemm_wmma_f16<_Float16, /*B_TRANS=*/true, /*BIAS=*/false, /*OUT_HALF=*/false>
        <<<dim3(TV / BN, TQ / BM, Bb), blk, 0, stream>>>(
            query, keys, nullptr, align,
            /*K=*/UN, /*lda=*/UN, /*ldb=*/UN, /*ldc=*/TV,
            (long long)TQ * UN, (long long)TV * UN, (long long)TQ * TV);

    // 3) alignment = softmax(score) in place
    softmax_rows_2048<<<dim3(Bb * TQ), blk, 0, stream>>>(align, TV);

    // 4) context = alignment @ values  (f32)
    gemm_wmma_f16<float, /*B_TRANS=*/false, /*BIAS=*/false, /*OUT_HALF=*/false>
        <<<dim3(Dd / BN, TQ / BM, Bb), blk, 0, stream>>>(
            align, values, nullptr, ctx,
            /*K=*/TV, /*lda=*/TV, /*ldb=*/Dd, /*ldc=*/Dd,
            (long long)TQ * TV, (long long)TV * Dd, (long long)TQ * Dd);
}
